// ConvSP_15367392985318
// MI455X (gfx1250) — matrix operand order repaired and checked by reading, not compiled, not run on hardware
//
#include <hip/hip_runtime.h>
#include <stdint.h>

typedef __attribute__((ext_vector_type(16))) __bf16 v16bf;
typedef __attribute__((ext_vector_type(8)))  float  v8f;

namespace {
constexpr int   kB      = 2;
constexpr int   kN      = 2048;
constexpr int   kCin    = 64;
constexpr int   kCout   = 64;
constexpr int   kCells  = 9;
constexpr int   kSplitK = 8;
constexpr int   kJChunk = 32;                 // K per bf16 WMMA
constexpr int   kNumJC  = kN / kJChunk;       // 64 chunks
constexpr float kR2     = 0.01f;              // RADIUS^2
constexpr float kKnorm  = (float)(315.0 / (64.0 * 3.14159265358979323846 * 1e-9));
constexpr size_t kPreElems = (size_t)kCells * kB * kCout * kN;    // bf16 elements
constexpr int   kTileB   = 512 * 2;           // bytes per 16x32 bf16 A tile
constexpr int   kStageB  = kCells * 4 * kTileB; // 36 KB: all A tiles of one chunk
}

// ---------------------------------------------------------------------------
// Kernel 1: pre[b][jc][k][otile] tile = knorm * sum_c weight[o,c,k]*data[b,c,j]
//           / (invmass*density), bf16, in exact WMMA A-operand per-lane order.
// Chunk-contiguous layout: the 36 tiles of one j-chunk form one 36 KB block so
// the main kernel's async fill is a single-base, immediate-offset clause.
// ---------------------------------------------------------------------------
__global__ void convsp_pre(const float* __restrict__ locs,
                           const float* __restrict__ data,
                           const float* __restrict__ density,
                           const float* __restrict__ weight,
                           __bf16* __restrict__ pre)
{
    const int tid = blockIdx.x * blockDim.x + threadIdx.x;
    const int j = tid & (kN - 1);
    const int o = (tid >> 11) & 63;
    const int b = (tid >> 17) & 1;
    const int k = tid >> 18;
    if (k >= kCells) return;

    const float invm = locs[((size_t)b * kN + j) * 3 + 2];
    const float dens = density[(size_t)b * kN + j];
    const float coef = kKnorm / (invm * dens);

    float acc = 0.0f;
    for (int c = 0; c < kCin; ++c) {
        const float wv = weight[((size_t)o * kCin + c) * kCells + k];
        const float dv = data[((size_t)b * kCin + c) * kN + j];
        acc = fmaf(wv, dv, acc);
    }
    acc *= coef;

    // (m = o%16, kk = j%32) -> (lane, elem) of the 16x32 bf16 A tile:
    //   e = 8*(kk/16) + kk%8 ; lane = 16*((kk/8)&1) + m
    const int otile = o >> 4, m = o & 15;
    const int jc = j >> 5, kk = j & 31;
    const int e    = 8 * (kk >> 4) + (kk & 7);
    const int lane = 16 * ((kk >> 3) & 1) + m;
    const size_t tile = (((size_t)b * kNumJC + jc) * kCells + k) * 4 + otile;
    pre[tile * 512 + (size_t)lane * 16 + e] = (__bf16)acc;
}

// ---------------------------------------------------------------------------
// Kernel 2: fused SPH-kernel-generation + GEMM via v_wmma_f32_16x16x32_bf16.
// A tiles double-buffered in LDS via CDNA5 async copies: one 36 KB contiguous
// block per chunk, filled by 18 global_load_async_to_lds_b128 issues per wave
// sharing one base address (INST_OFFSET applies to both LDS and global sides),
// drained with s_wait_asynccnt and overlapped with the WMMA compute.
// ---------------------------------------------------------------------------
__global__ void __launch_bounds__(128, 1)
convsp_main_wmma(const float* __restrict__ locs,
                 const __bf16* __restrict__ pre,
                 float* __restrict__ partial)
{
    const int blocksPerB = kN / 64;                       // 32
    const int s  = blockIdx.x / (kB * blocksPerB);        // split-K segment
    const int r  = blockIdx.x % (kB * blocksPerB);
    const int b  = r / blocksPerB;
    const int ig = r % blocksPerB;

    const int wave = threadIdx.x >> 5;
    const int lane = threadIdx.x & 31;
    const int n = lane & 15;
    const int h = lane >> 4;
    const int itile = ig * 4 + wave;
    const int i = itile * 16 + n;

    __shared__ __align__(32) uint8_t sA[2][kStageB];      // 72 KB double buffer
    __shared__ float sx[kJChunk];
    __shared__ float sy[kJChunk];

    const float pxi = locs[((size_t)b * kN + i) * 3 + 0];
    const float pyi = locs[((size_t)b * kN + i) * 3 + 1];

    v8f acc[4] = {v8f{}, v8f{}, v8f{}, v8f{}};

    const uint8_t* preB = (const uint8_t*)pre + (size_t)b * kNumJC * kStageB;
    const uint32_t sAoff = (uint32_t)(uintptr_t)(void*)&sA[0][0];
    const int jcPerSeg = kNumJC / kSplitK;                // 8

    // Async-fill buffer `buf` with chunk jc's 36 KB A-tile block.
    // 128 threads x 16 B x 18 rows; single base VGPRs, immediate row offsets.
    auto issue_copy = [&](int buf, int jc) {
        const uint64_t g = (uint64_t)(uintptr_t)
            (preB + (size_t)jc * kStageB + threadIdx.x * 16);
        const uint32_t l = sAoff + (uint32_t)(buf * kStageB) + threadIdx.x * 16;
#pragma unroll
        for (int row = 0; row < kStageB / 2048; ++row) {  // 18 issues
            asm volatile("global_load_async_to_lds_b128 %0, %1, off offset:%c2"
                         :: "v"(l), "v"(g), "i"(row * 2048) : "memory");
        }
    };

    issue_copy(0, s * jcPerSeg);                          // prologue fill

    for (int jj = 0; jj < jcPerSeg; ++jj) {
        const int jc  = s * jcPerSeg + jj;
        const int cur = jj & 1;

        asm volatile("s_wait_asynccnt 0x0" ::: "memory"); // this wave's copies done
        __syncthreads();                                  // LDS published; prev reads done

        if (jj + 1 < jcPerSeg)                            // overlap next fill with compute
            issue_copy(cur ^ 1, jc + 1);

        if (threadIdx.x < kJChunk) {                      // stage chunk positions
            const float* lp = &locs[((size_t)b * kN + jc * kJChunk + threadIdx.x) * 3];
            sx[threadIdx.x] = lp[0];
            sy[threadIdx.x] = lp[1];
        }
        __syncthreads();

        // pairwise deltas + squared distance, once per chunk (reused by all 9 cells)
        float dx[16], dy[16], s2[16];
#pragma unroll
        for (int e = 0; e < 16; ++e) {
            dx[e] = pxi - sx[16 * (e >> 3) + 8 * h + (e & 7)];
            dy[e] = pyi - sy[16 * (e >> 3) + 8 * h + (e & 7)];
            s2[e] = fmaf(dx[e], dx[e], dy[e] * dy[e]);
        }

#pragma unroll
        for (int k = 0; k < kCells; ++k) {
            constexpr float kd = 0.05f;
            const float ox = (float)(k / 3 - 1) * kd;     // compile-time per unrolled k
            const float oy = (float)(k % 3 - 1) * kd;
            const bool oxNZ = (k / 3) != 1;
            const bool oyNZ = (k % 3) != 1;
            const float c1 = -2.0f * ox;
            const float c2 = -2.0f * oy;
            const float c3 = kR2 - (ox * ox + oy * oy);

            v16bf bv;
#pragma unroll
            for (int e = 0; e < 16; ++e) {
                float t = c3 - s2[e];                     // r2 - |d+o|^2, expanded
                if (oxNZ) t = fmaf(c1, dx[e], t);
                if (oyNZ) t = fmaf(c2, dy[e], t);
                t = fmaxf(t, 0.0f);
                bv[e] = (__bf16)(t * t * t);              // knorm folded into A
            }

            const uint8_t* at = &sA[cur][(k * 4) * kTileB + lane * 32];
#pragma unroll
            for (int t4 = 0; t4 < 4; ++t4) {
                const v16bf av = *(const v16bf*)(at + t4 * kTileB);
                acc[t4] = __builtin_amdgcn_wmma_f32_16x16x32_bf16(
                    false, av, false, bv, (short)0, acc[t4], false, false);
            }
        }
        __syncthreads();                                  // reads done before buffer reuse
    }

    // C/D layout: VGPR rr, lane-half h -> row m = rr + 8h, col = n
    float* pout = partial + ((size_t)s * kB + b) * kCout * kN;
#pragma unroll
    for (int t4 = 0; t4 < 4; ++t4) {
#pragma unroll
        for (int rr = 0; rr < 8; ++rr) {
            const int o = t4 * 16 + rr + 8 * h;
            pout[(size_t)o * kN + i] = acc[t4][rr];
        }
    }
}

// ---------------------------------------------------------------------------
// Kernel 3: out = bias + sum over split-K partials (deterministic reduction).
// ---------------------------------------------------------------------------
__global__ void convsp_reduce(const float* __restrict__ partial,
                              const float* __restrict__ bias,
                              float* __restrict__ out, int total)
{
    const int idx = blockIdx.x * blockDim.x + threadIdx.x;
    if (idx >= total) return;
    const int o = (idx / kN) % kCout;
    float a = bias[o];
    for (int s = 0; s < kSplitK; ++s)
        a += partial[(size_t)s * kB * kCout * kN + idx];
    out[idx] = a;
}

// ---------------------------------------------------------------------------
extern "C" void kernel_launch(void* const* d_in, const int* in_sizes, int n_in,
                              void* d_out, int out_size, void* d_ws, size_t ws_size,
                              hipStream_t stream)
{
    const float* locs    = (const float*)d_in[0];
    const float* data    = (const float*)d_in[1];
    const float* density = (const float*)d_in[2];
    const float* weight  = (const float*)d_in[3];
    const float* bias    = (const float*)d_in[4];
    float* out = (float*)d_out;

    __bf16* pre    = (__bf16*)d_ws;
    float* partial = (float*)((char*)d_ws + kPreElems * sizeof(__bf16));

    {   // (k,b,o,j) one thread each
        const int total = kCells * kB * kCout * kN;       // 9<<18
        convsp_pre<<<total / 256, 256, 0, stream>>>(locs, data, density, weight, pre);
    }
    {
        dim3 grid(kSplitK * kB * (kN / 64));              // 512 workgroups
        convsp_main_wmma<<<grid, 128, 0, stream>>>(locs, pre, partial);
    }
    {
        const int total = kB * kCout * kN;                // 262144
        convsp_reduce<<<(total + 255) / 256, 256, 0, stream>>>(partial, bias, out, total);
    }
}
